// MultiHeadAttention_40716289966411
// MI455X (gfx1250) — matrix operand order repaired and checked by reading, not compiled, not run on hardware
//
#include <hip/hip_runtime.h>

// ---------------------------------------------------------------------------
// MI455X (gfx1250) multi-head attention forward.
// bf16 operands + f32 WMMA accumulate (v_wmma_f32_16x16x32_bf16), wave32.
// Generic C = alpha*(A @ B^T) + bias WMMA kernel, 32x64 wave tile,
// ping-pong LDS B staging via async global->LDS loads (ASYNCcnt) when
// available, single barrier per k-step.
// ---------------------------------------------------------------------------

typedef __attribute__((ext_vector_type(16))) __bf16 v16bf;
typedef __attribute__((ext_vector_type(8)))  __bf16 v8bf;
typedef __attribute__((ext_vector_type(4)))  __bf16 v4bf;
typedef __attribute__((ext_vector_type(8)))  float  v8f;
typedef __attribute__((ext_vector_type(4)))  float  v4f;
typedef __attribute__((ext_vector_type(4)))  int    v4i;

#define NB      2
#define SEQ     2048
#define DMODEL  1024
#define NHEAD   16
#define DK      64

#if __has_builtin(__builtin_amdgcn_global_load_async_to_lds_b128)
#define HAVE_ASYNC_LDS 1
#else
#define HAVE_ASYNC_LDS 0
#endif

typedef __attribute__((address_space(1))) v4i* gbl_v4i_p;
typedef __attribute__((address_space(3))) v4i* lds_v4i_p;

__device__ __forceinline__ void async_wait0() {
#if HAVE_ASYNC_LDS
#if __has_builtin(__builtin_amdgcn_s_wait_asynccnt)
  __builtin_amdgcn_s_wait_asynccnt(0);
#else
  asm volatile("s_wait_asynccnt 0x0" ::: "memory");
#endif
#endif
}

// Stage 16 contiguous bf16 (32B) from global into LDS.
__device__ __forceinline__ void stage16(const __bf16* __restrict__ g,
                                        __bf16* __restrict__ l) {
#if HAVE_ASYNC_LDS
  __bf16* gnc = const_cast<__bf16*>(g);
  __builtin_amdgcn_global_load_async_to_lds_b128(
      (gbl_v4i_p)gnc, (lds_v4i_p)l, 0, 0);
  __builtin_amdgcn_global_load_async_to_lds_b128(
      (gbl_v4i_p)(gnc + 8), (lds_v4i_p)(l + 8), 0, 0);
#else
  v8bf b0 = *(const v8bf*)g;
  v8bf b1 = *(const v8bf*)(g + 8);
  *(v8bf*)l = b0;
  *(v8bf*)(l + 8) = b1;
#endif
}

// ---------------------------------------------------------------------------
// f32 -> bf16 cast (vector-of-4 per thread)
// ---------------------------------------------------------------------------
__global__ __launch_bounds__(256) void cast_f32_bf16(const float* __restrict__ in,
                                                     __bf16* __restrict__ out, int n4) {
  int i = blockIdx.x * 256 + threadIdx.x;
  if (i < n4) {
    v4f x = ((const v4f*)in)[i];
    v4bf y{};
#pragma unroll
    for (int j = 0; j < 4; ++j) y[j] = (__bf16)x[j];
    ((v4bf*)out)[i] = y;
  }
}

// ---------------------------------------------------------------------------
// Per-(b,h) transpose of V projection: [S, dk] (ld DMODEL) -> vT [dk, S]
// grid: (SEQ/64, NB*NHEAD), block 256
// ---------------------------------------------------------------------------
__global__ __launch_bounds__(256) void transpose_v_kernel(const __bf16* __restrict__ vp,
                                                          __bf16* __restrict__ vT) {
  const int bh = blockIdx.y;
  const int b  = bh >> 4;
  const int h  = bh & 15;
  const int s0 = blockIdx.x * 64;
  __shared__ __bf16 tile[64][72];

  const int t  = threadIdx.x;
  const int r  = t >> 2;           // 0..63
  const int c0 = (t & 3) * 16;     // 0,16,32,48
  {
    const __bf16* src = vp + (long)(b * SEQ + s0 + r) * DMODEL + h * DK + c0;
    v8bf x0 = *(const v8bf*)src;
    v8bf x1 = *(const v8bf*)(src + 8);
    *(v8bf*)&tile[r][c0] = x0;
    *(v8bf*)&tile[r][c0 + 8] = x1;
  }
  __syncthreads();
  {
    const int d = r;  // 0..63
    v8bf y0{}, y1{};
#pragma unroll
    for (int j = 0; j < 8; ++j) { y0[j] = tile[c0 + j][d]; y1[j] = tile[c0 + 8 + j][d]; }
    __bf16* dst = vT + ((long)(bh * DK + d)) * SEQ + s0 + c0;
    *(v8bf*)dst = y0;
    *(v8bf*)(dst + 8) = y1;
  }
}

// ---------------------------------------------------------------------------
// Row softmax over S=2048 elements, one block (256 thr, 8 elem/thr) per row.
// ---------------------------------------------------------------------------
__global__ __launch_bounds__(256) void softmax_rows(float* __restrict__ attn) {
  float* p = attn + (long)blockIdx.x * SEQ;
  const int t = threadIdx.x;
  v4f x0 = ((const v4f*)p)[t * 2];
  v4f x1 = ((const v4f*)p)[t * 2 + 1];

  __shared__ float red[256];
  float m = -3.4e38f;
#pragma unroll
  for (int j = 0; j < 4; ++j) { m = fmaxf(m, x0[j]); m = fmaxf(m, x1[j]); }
  red[t] = m;
  __syncthreads();
#pragma unroll
  for (int s = 128; s > 0; s >>= 1) {
    if (t < s) red[t] = fmaxf(red[t], red[t + s]);
    __syncthreads();
  }
  m = red[0];
  __syncthreads();

  float sum = 0.f;
#pragma unroll
  for (int j = 0; j < 4; ++j) {
    x0[j] = __expf(x0[j] - m); sum += x0[j];
    x1[j] = __expf(x1[j] - m); sum += x1[j];
  }
  red[t] = sum;
  __syncthreads();
#pragma unroll
  for (int s = 128; s > 0; s >>= 1) {
    if (t < s) red[t] += red[t + s];
    __syncthreads();
  }
  const float inv = 1.f / red[0];
#pragma unroll
  for (int j = 0; j < 4; ++j) { x0[j] *= inv; x1[j] *= inv; }
  ((v4f*)p)[t * 2]     = x0;
  ((v4f*)p)[t * 2 + 1] = x1;
}

// ---------------------------------------------------------------------------
// Generic WMMA GEMM: C = alpha*(A @ B^T) (+ bias[n]).
// A: [M,K] (bf16, or f32 converted on the fly), B: [N,K] bf16, K-contiguous.
// Block = 128 threads = 4 waves; block tile 128(M)x64(N); wave tile 32x64.
// B tile (64x32 bf16) ping-pong staged in LDS; one barrier per k-step.
// ---------------------------------------------------------------------------
template <int A_IS_F32, int C_IS_BF16, int HAS_BIAS>
__global__ __launch_bounds__(128) void wmma_gemm_bt(
    const void* __restrict__ Ap, const __bf16* __restrict__ Bp,
    const float* __restrict__ bias, void* __restrict__ Cp,
    int M, int N, int K, int lda, int ldb, int ldc, int nH,
    long sAb, long sAh, long sBb, long sBh, long sCb, long sCh, float alpha) {
  const int z  = blockIdx.z;
  const int bb = z / nH;
  const int hh = z % nH;
  const long aBatch = (long)bb * sAb + (long)hh * sAh;
  const long bBatch = (long)bb * sBb + (long)hh * sBh;
  const long cBatch = (long)bb * sCb + (long)hh * sCh;

  const int lane  = threadIdx.x & 31;
  const int wave  = threadIdx.x >> 5;
  const int mBase = blockIdx.y * 128 + wave * 32;
  const int nBase = blockIdx.x * 64;

  __shared__ __bf16 sB[2][64 * 40];  // 64 N-rows x 32 K, padded stride 40 (80B)

  const int am0  = mBase + (lane & 15);       // A frag 0 rows
  const int am1  = am0 + 16;                  // A frag 1 rows
  const int kSel = (lane >> 4) << 3;          // A: K chunk select (ISA layout)
  const int bRow = lane & 15;                 // B: column n = lane%16
  const int bkSel = (lane >> 4) << 3;         // B: K chunk select (ISA layout, matches A)

  // cooperative B-tile staging addressing: 128 thr x 16 bf16 = 64x32 tile
  const int sr = threadIdx.x >> 1;            // 0..63 (N row)
  const int sc = (threadIdx.x & 1) * 16;      // 0 or 16 (K col)
  const __bf16* bSrc = Bp + bBatch + (long)(nBase + sr) * ldb + sc;

  v8f acc[2][4] = {{v8f{}, v8f{}, v8f{}, v8f{}}, {v8f{}, v8f{}, v8f{}, v8f{}}};

  const int T = K >> 5;  // k-steps of 32

  // prologue: stage tile 0 into buffer 0
  stage16(bSrc, &sB[0][sr * 40 + sc]);

  for (int i = 0; i < T; ++i) {
    async_wait0();      // async loads for buf[i&1] complete (issued at i-1)
    __syncthreads();    // visible to all waves; prior reads of buf[(i+1)&1] done

    if (i + 1 < T)      // stage next tile into the other buffer (overlaps WMMAs)
      stage16(bSrc + (i + 1) * 32, &sB[(i + 1) & 1][sr * 40 + sc]);

    const int kk = i << 5;

    // ---- two A fragments (documented 16-bit A 16x32 VGPR layout) ----
    v16bf a0, a1;
    if (A_IS_F32) {
      const float* ap0 = (const float*)Ap + aBatch + (long)am0 * lda + kk + kSel;
      const float* ap1 = (const float*)Ap + aBatch + (long)am1 * lda + kk + kSel;
      v4f f[8];
      f[0] = *(const v4f*)(ap0);      f[1] = *(const v4f*)(ap0 + 4);
      f[2] = *(const v4f*)(ap0 + 16); f[3] = *(const v4f*)(ap0 + 20);
      f[4] = *(const v4f*)(ap1);      f[5] = *(const v4f*)(ap1 + 4);
      f[6] = *(const v4f*)(ap1 + 16); f[7] = *(const v4f*)(ap1 + 20);
      v16bf t0{}, t1{};
#pragma unroll
      for (int j = 0; j < 4; ++j) {
        t0[j] = (__bf16)f[0][j]; t0[j + 4]  = (__bf16)f[1][j];
        t0[j + 8] = (__bf16)f[2][j]; t0[j + 12] = (__bf16)f[3][j];
        t1[j] = (__bf16)f[4][j]; t1[j + 4]  = (__bf16)f[5][j];
        t1[j + 8] = (__bf16)f[6][j]; t1[j + 12] = (__bf16)f[7][j];
      }
      a0 = t0; a1 = t1;
    } else {
      const __bf16* ap0 = (const __bf16*)Ap + aBatch + (long)am0 * lda + kk + kSel;
      const __bf16* ap1 = (const __bf16*)Ap + aBatch + (long)am1 * lda + kk + kSel;
      v8bf lo0 = *(const v8bf*)(ap0), hi0 = *(const v8bf*)(ap0 + 16);
      v8bf lo1 = *(const v8bf*)(ap1), hi1 = *(const v8bf*)(ap1 + 16);
      a0 = __builtin_shufflevector(lo0, hi0, 0, 1, 2, 3, 4, 5, 6, 7,
                                   8, 9, 10, 11, 12, 13, 14, 15);
      a1 = __builtin_shufflevector(lo1, hi1, 0, 1, 2, 3, 4, 5, 6, 7,
                                   8, 9, 10, 11, 12, 13, 14, 15);
    }

    // ---- load all 4 B fragments, then 8 back-to-back WMMAs ----
    const __bf16* sbuf = &sB[i & 1][0];
    v16bf bf[4];
#pragma unroll
    for (int f = 0; f < 4; ++f) {
      const __bf16* bp = sbuf + (f * 16 + bRow) * 40 + bkSel;
      v8bf blo = *(const v8bf*)(bp);
      v8bf bhi = *(const v8bf*)(bp + 16);
      bf[f] = __builtin_shufflevector(blo, bhi, 0, 1, 2, 3, 4, 5, 6, 7,
                                      8, 9, 10, 11, 12, 13, 14, 15);
    }
#pragma unroll
    for (int f = 0; f < 4; ++f) {
      acc[0][f] = __builtin_amdgcn_wmma_f32_16x16x32_bf16(
          false, a0, false, bf[f], (short)0, acc[0][f], false, false);
      acc[1][f] = __builtin_amdgcn_wmma_f32_16x16x32_bf16(
          false, a1, false, bf[f], (short)0, acc[1][f], false, false);
    }
  }

  // ---- epilogue: row = mBase + m*16 + (lane>=16?8:0) + r, col = lane&15 ----
  const int cmHalf = (lane >> 4) << 3;
  const int cn = lane & 15;
#pragma unroll
  for (int mi = 0; mi < 2; ++mi) {
    const int cm = mBase + mi * 16 + cmHalf;
#pragma unroll
    for (int f = 0; f < 4; ++f) {
      const int n = nBase + f * 16 + cn;
      const float bv = HAS_BIAS ? bias[n] : 0.f;
#pragma unroll
      for (int r = 0; r < 8; ++r) {
        const float v = acc[mi][f][r] * alpha + bv;
        const long idx = cBatch + (long)(cm + r) * ldc + n;
        if (C_IS_BF16)
          ((__bf16*)Cp)[idx] = (__bf16)v;
        else
          ((float*)Cp)[idx] = v;
      }
    }
  }
}

// ---------------------------------------------------------------------------
// Launcher
// ---------------------------------------------------------------------------
extern "C" void kernel_launch(void* const* d_in, const int* in_sizes, int n_in,
                              void* d_out, int out_size, void* d_ws, size_t ws_size,
                              hipStream_t stream) {
  const float* Q   = (const float*)d_in[0];
  const float* K   = (const float*)d_in[1];
  const float* V   = (const float*)d_in[2];
  const float* Wq  = (const float*)d_in[3];
  const float* bq  = (const float*)d_in[4];
  const float* Wk  = (const float*)d_in[5];
  const float* bk  = (const float*)d_in[6];
  const float* Wv  = (const float*)d_in[7];
  const float* bv  = (const float*)d_in[8];
  const float* Wfc = (const float*)d_in[9];
  const float* bfc = (const float*)d_in[10];

  const long NE = (long)NB * SEQ * DMODEL;  // 4,194,304
  const long WE = (long)DMODEL * DMODEL;    // 1,048,576

  __bf16* ws   = (__bf16*)d_ws;
  __bf16* Qb   = ws;
  __bf16* Kb   = Qb + NE;
  __bf16* Vb   = Kb + NE;
  __bf16* Wqb  = Vb + NE;
  __bf16* Wkb  = Wqb + WE;
  __bf16* Wvb  = Wkb + WE;
  __bf16* Wfcb = Wvb + WE;
  __bf16* qp   = Wfcb + WE;
  __bf16* kp   = qp + NE;
  __bf16* vp   = kp + NE;
  __bf16* vT   = vp + NE;   // [NB*NHEAD, DK, SEQ] = NE elements
  __bf16* ctx  = vT + NE;   // total 9*NE bf16 = 72 MB

  float* out  = (float*)d_out;                 // [NB,SEQ,DMODEL]
  float* attn = out + NE;                      // [NB,NHEAD,SEQ,SEQ]

  // 1) casts to bf16
  cast_f32_bf16<<<(int)(NE / 4 / 256), 256, 0, stream>>>(Q, Qb, (int)(NE / 4));
  cast_f32_bf16<<<(int)(NE / 4 / 256), 256, 0, stream>>>(K, Kb, (int)(NE / 4));
  cast_f32_bf16<<<(int)(NE / 4 / 256), 256, 0, stream>>>(V, Vb, (int)(NE / 4));
  cast_f32_bf16<<<(int)(WE / 4 / 256), 256, 0, stream>>>(Wq, Wqb, (int)(WE / 4));
  cast_f32_bf16<<<(int)(WE / 4 / 256), 256, 0, stream>>>(Wk, Wkb, (int)(WE / 4));
  cast_f32_bf16<<<(int)(WE / 4 / 256), 256, 0, stream>>>(Wv, Wvb, (int)(WE / 4));
  cast_f32_bf16<<<(int)(WE / 4 / 256), 256, 0, stream>>>(Wfc, Wfcb, (int)(WE / 4));

  // 2) QKV projections: [4096,1024] @ [1024,1024]^T + bias -> bf16
  {
    dim3 g(DMODEL / 64, (NB * SEQ) / 128, 1);
    wmma_gemm_bt<0, 1, 1><<<g, 128, 0, stream>>>(Qb, Wqb, bq, qp,
        NB * SEQ, DMODEL, DMODEL, DMODEL, DMODEL, DMODEL,
        1, 0, 0, 0, 0, 0, 0, 1.0f);
    wmma_gemm_bt<0, 1, 1><<<g, 128, 0, stream>>>(Kb, Wkb, bk, kp,
        NB * SEQ, DMODEL, DMODEL, DMODEL, DMODEL, DMODEL,
        1, 0, 0, 0, 0, 0, 0, 1.0f);
    wmma_gemm_bt<0, 1, 1><<<g, 128, 0, stream>>>(Vb, Wvb, bv, vp,
        NB * SEQ, DMODEL, DMODEL, DMODEL, DMODEL, DMODEL,
        1, 0, 0, 0, 0, 0, 0, 1.0f);
  }

  // 3) V^T per head: [S,dk] -> [dk,S]
  transpose_v_kernel<<<dim3(SEQ / 64, NB * NHEAD), 256, 0, stream>>>(vp, vT);

  // 4) scores = (q @ k^T) / 8 -> attn region (f32), batched over (b,h)
  wmma_gemm_bt<0, 0, 0><<<dim3(SEQ / 64, SEQ / 128, NB * NHEAD), 128, 0, stream>>>(
      qp, kp, nullptr, attn,
      SEQ, SEQ, DK, DMODEL, DMODEL, SEQ, NHEAD,
      (long)SEQ * DMODEL, DK,                     // A strides (b, h)
      (long)SEQ * DMODEL, DK,                     // B strides
      (long)NHEAD * SEQ * SEQ, (long)SEQ * SEQ,   // C strides
      0.125f);

  // 5) softmax rows (in-place in d_out attn region)
  softmax_rows<<<NB * NHEAD * SEQ, 256, 0, stream>>>(attn);

  // 6) ctx = attn @ vT^T  (A = f32 attn converted on load) -> bf16 ctx buffer
  wmma_gemm_bt<1, 1, 0><<<dim3(DK / 64, SEQ / 128, NB * NHEAD), 128, 0, stream>>>(
      attn, vT, nullptr, ctx,
      SEQ, DK, SEQ, SEQ, SEQ, DMODEL, NHEAD,
      (long)NHEAD * SEQ * SEQ, (long)SEQ * SEQ,   // A strides
      (long)NHEAD * DK * SEQ, (long)DK * SEQ,     // B strides
      (long)SEQ * DMODEL, DK,                     // C strides (head -> col offset)
      1.0f);

  // 7) out = ctx @ Wfc^T + bfc -> f32 d_out
  wmma_gemm_bt<0, 0, 1><<<dim3(DMODEL / 64, (NB * SEQ) / 128, 1), 128, 0, stream>>>(
      ctx, Wfcb, bfc, out,
      NB * SEQ, DMODEL, DMODEL, DMODEL, DMODEL, DMODEL,
      1, 0, 0, 0, 0, 0, 0, 1.0f);
}
